// GraphNetwork_RGCN_962072674435
// MI455X (gfx1250) — compile-verified, matrix operand order checked by reading, not run on hardware
//
#include <hip/hip_runtime.h>
#include <hip/hip_bf16.h>

// ---------------------------------------------------------------------------
// MI455X (gfx1250, wave32) implementation.
//
// Strategy (roofline): fold the per-relation RGCN transform + root weight into
// ONE dense GEMM  Hcat[N,1088] = x[N,256] @ [W_0|...|W_15|root] (f16 in, f32
// acc via v_wmma_f32_16x16x32_f16, f16 out to halve HBM traffic).  Edge
// messages are then 128B contiguous gathers Hcat[src, et*64..] (HBM-latency
// bound -> software prefetch via global_prefetch_b8).  cnt/agg/agg2/h1
// (<=26MB each) are L2-resident (192MB L2) so all atomics hit L2.
// GraphConv is a second f16 WMMA GEMM: out = [agg2|h1] @ [w_rel;w_root] + b.
// GEMM wave tile = 32x64 (8 accumulators): 12 b128 loads per 8 WMMAs.
// ---------------------------------------------------------------------------

#define NNODES 100000
#define NEDGES 1600000
#define FIN    256
#define HID    64
#define NREL   16
#define NBASES 30
#define RC     (NREL*HID + HID)   // 1088 columns of GEMM1 output

typedef __attribute__((ext_vector_type(16))) _Float16 v16h;
typedef __attribute__((ext_vector_type(8)))  _Float16 v8h;
typedef __attribute__((ext_vector_type(8)))  float    v8f;

#define SHUF16(lo, hi) __builtin_shufflevector(lo, hi, 0,1,2,3,4,5,6,7,8,9,10,11,12,13,14,15)

// ---- x (f32) -> f16 ----
__global__ void k_x2h(const float* __restrict__ x, _Float16* __restrict__ xh, int n) {
  int i = blockIdx.x * 256 + threadIdx.x;
  if (i < n) xh[i] = (_Float16)x[i];
}

// ---- build B^T for GEMM1: Wt[c][k], c<1024 -> W[r=c>>6][k][o=c&63], else root ----
__global__ void k_build_w(const float* __restrict__ comp, const float* __restrict__ basis,
                          const float* __restrict__ root, _Float16* __restrict__ Wt) {
  int i = blockIdx.x * 256 + threadIdx.x;        // RC*FIN threads
  if (i >= RC * FIN) return;
  int c = i / FIN, k = i - c * FIN;
  float acc;
  if (c < NREL * HID) {
    int r = c >> 6, o = c & 63;
    acc = 0.f;
    for (int b = 0; b < NBASES; ++b)
      acc += comp[r * NBASES + b] * basis[((size_t)b * FIN + k) * HID + o];
  } else {
    acc = root[k * HID + (c - NREL * HID)];
  }
  Wt[(size_t)c * FIN + k] = (_Float16)acc;
}

// ---- build B^T for GEMM2: Wt2[c][k]: k<64 -> w_rel[k][c], else w_root[k-64][c] ----
__global__ void k_build_w2(const float* __restrict__ w_rel, const float* __restrict__ w_root,
                           _Float16* __restrict__ Wt2) {
  int i = blockIdx.x * 256 + threadIdx.x;        // 64*128 threads
  if (i >= HID * 2 * HID) return;
  int c = i >> 7, k = i & 127;
  float v = (k < HID) ? w_rel[k * HID + c] : w_root[(k - HID) * HID + c];
  Wt2[c * (2 * HID) + k] = (_Float16)v;
}

__global__ void k_zero32(unsigned* __restrict__ p, int n) {
  int i = blockIdx.x * 256 + threadIdx.x;
  if (i < n) p[i] = 0u;
}

// ---- per-(dst,relation) in-degree histogram ----
__global__ void k_count(const int* __restrict__ ei, const int* __restrict__ et,
                        unsigned* __restrict__ cnt) {
  int e = blockIdx.x * 256 + threadIdx.x;
  if (e >= NEDGES) return;
  int dst = ei[NEDGES + e];
  atomicAdd(&cnt[dst * NREL + et[e]], 1u);
}

// ---- generic f16 WMMA GEMM: C[M, ncols] = A[M,K] @ Bt[ncols,K]^T
//      one wave -> 32x64 C tile (2 A-tiles x 4 B-tiles, 8 accumulators) ----
template <int K, bool OUT_F16>
__global__ void k_gemm(const _Float16* __restrict__ A, const _Float16* __restrict__ Bt,
                       void* __restrict__ Cout, const float* __restrict__ bias,
                       int M, int ldc) {
  const int lane = threadIdx.x & 31;
  const int wave = threadIdx.x >> 5;
  const int mt2  = blockIdx.x * 8 + wave;        // 32-row tile index
  if (mt2 * 32 >= M) return;
  const int cg   = blockIdx.y;                   // 64-col group
  const int rn   = lane & 15;
  const int hsel = lane >> 4;                    // 0: K 0-7/16-23, 1: K 8-15/24-31

  const _Float16* arow0 = A + (size_t)(mt2 * 32 + rn) * K;
  const _Float16* arow1 = arow0 + (size_t)16 * K;
  const _Float16* brow0 = Bt + (size_t)(cg * 64 + rn) * K;

  v8f acc0[4] = {};
  v8f acc1[4] = {};
#pragma unroll
  for (int kc = 0; kc < K; kc += 32) {
    v8h a0lo = *(const v8h*)(arow0 + kc + hsel * 8);
    v8h a0hi = *(const v8h*)(arow0 + kc + 16 + hsel * 8);
    v8h a1lo = *(const v8h*)(arow1 + kc + hsel * 8);
    v8h a1hi = *(const v8h*)(arow1 + kc + 16 + hsel * 8);
    v16h a0 = SHUF16(a0lo, a0hi);
    v16h a1 = SHUF16(a1lo, a1hi);
#pragma unroll
    for (int t = 0; t < 4; ++t) {
      const _Float16* brow = brow0 + (size_t)t * 16 * K;
      v8h blo = *(const v8h*)(brow + kc + hsel * 8);
      v8h bhi = *(const v8h*)(brow + kc + 16 + hsel * 8);
      v16h b  = SHUF16(blo, bhi);
      acc0[t] = __builtin_amdgcn_wmma_f32_16x16x32_f16(
          false, a0, false, b, (short)0, acc0[t], false, false);
      acc1[t] = __builtin_amdgcn_wmma_f32_16x16x32_f16(
          false, a1, false, b, (short)0, acc1[t], false, false);
    }
  }

  // C/D layout: VGPR v -> M = v + 8*hsel; N = lane&15
#pragma unroll
  for (int t = 0; t < 4; ++t) {
    int coln = cg * 64 + t * 16 + rn;
#pragma unroll
    for (int v = 0; v < 8; ++v) {
      int m0 = mt2 * 32 + v + 8 * hsel;
      int m1 = m0 + 16;
      if (OUT_F16) {
        ((_Float16*)Cout)[(size_t)m0 * ldc + coln] = (_Float16)acc0[t][v];
        ((_Float16*)Cout)[(size_t)m1 * ldc + coln] = (_Float16)acc1[t][v];
      } else {
        float bb = bias[coln];
        ((float*)Cout)[(size_t)m0 * ldc + coln] = acc0[t][v] + bb;
        ((float*)Cout)[(size_t)m1 * ldc + coln] = acc1[t][v] + bb;
      }
    }
  }
}

// ---- RGCN edge scatter: wave per edge, lane handles 2 of 64 features.
//      Hcat (218MB) misses L2 -> prefetch the gather target PF edges ahead. ----
#define PF_DIST 64
__global__ void k_scatter1(const int* __restrict__ ei, const int* __restrict__ et,
                           const _Float16* __restrict__ Hcat, const unsigned* __restrict__ cnt,
                           float* __restrict__ agg) {
  int lane = threadIdx.x & 31;
  int e = (blockIdx.x * 256 + threadIdx.x) >> 5;
  if (e >= NEDGES) return;

  int ep = e + PF_DIST;
  if (ep < NEDGES) {
    int srcp = ei[ep], rp = et[ep];
    // each lane prefetches its 4B slice of the future 128B gather segment
    __builtin_prefetch(Hcat + (size_t)srcp * RC + rp * HID + lane * 2, 0, 0);
  }

  int src = ei[e], dst = ei[NEDGES + e], r = et[e];
  unsigned c = cnt[dst * NREL + r];
  float coeff = 1.0f / (float)(c ? c : 1u);
  const _Float16* p = Hcat + (size_t)src * RC + r * HID + lane * 2;
  float m0 = (float)p[0] * coeff;
  float m1 = (float)p[1] * coeff;
  atomicAdd(&agg[dst * HID + lane * 2],     m0);
  atomicAdd(&agg[dst * HID + lane * 2 + 1], m1);
}

// ---- h1 = agg + (x@root) + bias1 ; also stash f16 copy into AB[:,64:128] ----
__global__ void k_h1(const float* __restrict__ agg, const _Float16* __restrict__ Hcat,
                     const float* __restrict__ bias1, float* __restrict__ h1,
                     _Float16* __restrict__ AB) {
  int i = blockIdx.x * 256 + threadIdx.x;        // N*HID
  if (i >= NNODES * HID) return;
  int n = i >> 6, o = i & 63;
  float v = agg[i] + (float)Hcat[(size_t)n * RC + NREL * HID + o] + bias1[o];
  h1[i] = v;
  AB[(size_t)n * 128 + 64 + o] = (_Float16)v;
}

// ---- GraphConv edge scatter: agg2[dst] += h1[src] (h1 is L2-resident) ----
__global__ void k_scatter2(const int* __restrict__ ei, const float* __restrict__ h1,
                           float* __restrict__ agg2) {
  int lane = threadIdx.x & 31;
  int e = (blockIdx.x * 256 + threadIdx.x) >> 5;
  if (e >= NEDGES) return;
  int src = ei[e], dst = ei[NEDGES + e];
  float2 v = *((const float2*)(h1 + (size_t)src * HID) + lane);
  atomicAdd(&agg2[dst * HID + lane * 2],     v.x);
  atomicAdd(&agg2[dst * HID + lane * 2 + 1], v.y);
}

// ---- agg2 -> f16 into AB[:,0:64] ----
__global__ void k_ab(const float* __restrict__ agg2, _Float16* __restrict__ AB) {
  int i = blockIdx.x * 256 + threadIdx.x;
  if (i >= NNODES * HID) return;
  int n = i >> 6, o = i & 63;
  AB[(size_t)n * 128 + o] = (_Float16)agg2[i];
}

// ---------------------------------------------------------------------------
extern "C" void kernel_launch(void* const* d_in, const int* in_sizes, int n_in,
                              void* d_out, int out_size, void* d_ws, size_t ws_size,
                              hipStream_t stream) {
  const float* x      = (const float*)d_in[0];
  const int*   ei     = (const int*)d_in[1];
  const int*   et     = (const int*)d_in[2];
  const float* comp   = (const float*)d_in[3];
  const float* basis  = (const float*)d_in[4];
  const float* root   = (const float*)d_in[5];
  const float* bias1  = (const float*)d_in[6];
  const float* w_rel  = (const float*)d_in[7];
  const float* b_rel  = (const float*)d_in[8];
  const float* w_root = (const float*)d_in[9];
  float* out = (float*)d_out;

  char* ws = (char*)d_ws;
  // workspace layout (all 256B aligned)
  _Float16* XH   = (_Float16*)(ws);                                   //  51,200,000 B
  _Float16* WT   = (_Float16*)(ws + 51200000);                        //     557,056 B
  _Float16* WT2  = (_Float16*)(ws + 51757056);                        //      16,384 B
  _Float16* HC   = (_Float16*)(ws + 51773440);                        // 217,600,000 B
  unsigned* CNT  = (unsigned*)(ws + 269373440);                       //   6,400,000 B
  float*    AGG  = (float*)   (ws + 275773440);                       //  25,600,000 B
  float*    AGG2 = (float*)   (ws + 301373440);                       //  25,600,000 B
  float*    H1   = (float*)   (ws + 326973440);                       //  25,600,000 B
  _Float16* AB   = (_Float16*)(ws + 352573440);                       //  25,600,000 B

  const int nXF = NNODES * FIN;               // 25.6M
  const int nNH = NNODES * HID;               // 6.4M
  const int nZero = NNODES * NREL + 2 * nNH;  // cnt + agg + agg2 (contiguous u32 span)

  k_x2h<<<(nXF + 255) / 256, 256, 0, stream>>>(x, XH, nXF);
  k_build_w<<<(RC * FIN + 255) / 256, 256, 0, stream>>>(comp, basis, root, WT);
  k_build_w2<<<(HID * 2 * HID + 255) / 256, 256, 0, stream>>>(w_rel, w_root, WT2);
  k_zero32<<<(nZero + 255) / 256, 256, 0, stream>>>(CNT, nZero);

  // GEMM1: Hcat[N,1088] = XH @ WT^T   (f16 out)
  {
    int mtiles = NNODES / 32;                       // 3125
    dim3 grid((mtiles + 7) / 8, RC / 64);           // (391, 17)
    k_gemm<FIN, true><<<grid, 256, 0, stream>>>(XH, WT, HC, nullptr, NNODES, RC);
  }

  k_count<<<(NEDGES + 255) / 256, 256, 0, stream>>>(ei, et, CNT);
  k_scatter1<<<NEDGES / 8, 256, 0, stream>>>(ei, et, HC, CNT, AGG);
  k_h1<<<(nNH + 255) / 256, 256, 0, stream>>>(AGG, HC, bias1, H1, AB);
  k_scatter2<<<NEDGES / 8, 256, 0, stream>>>(ei, H1, AGG2);
  k_ab<<<(nNH + 255) / 256, 256, 0, stream>>>(AGG2, AB);

  // GEMM2: out[N,64] = AB[N,128] @ WT2^T + b_rel   (f32 out)
  {
    int mtiles = NNODES / 32;
    dim3 grid((mtiles + 7) / 8, 1);
    k_gemm<2 * HID, false><<<grid, 256, 0, stream>>>(AB, WT2, out, b_rel, NNODES, HID);
  }

  (void)in_sizes; (void)n_in; (void)out_size; (void)ws_size;
}
